// Encoder_60438779789634
// MI455X (gfx1250) — matrix operand order repaired and checked by reading, not compiled, not run on hardware
//
#include <hip/hip_runtime.h>
#include <cmath>

// CDNA5 / gfx1250, wave32. WMMA bf16 16x16x32 with f32 accumulation.
typedef __attribute__((ext_vector_type(16))) __bf16         v16bf;
typedef __attribute__((ext_vector_type(8)))  float          v8f;
typedef __attribute__((ext_vector_type(16))) unsigned short v16u;

#define B_ 32
#define T_ 512
#define E_ 256
#define U_ 256
#define G_ 1024            // 4*U
#define V_ 50257
#define MROWS (B_ * T_)    // 16384
#define FRAG_ELEMS (64 * 8 * 32 * 16)  // nt(64) x kt(8) x lane(32) x e(16) = 262144

__device__ __forceinline__ unsigned short f2bf(float f) {
    union { float f; unsigned u; } x; x.f = f;
    unsigned u = x.u;
    u += 0x7FFFu + ((u >> 16) & 1u);   // round-to-nearest-even
    return (unsigned short)(u >> 16);
}
__device__ __forceinline__ float sigmoidf(float x) { return 1.0f / (1.0f + __expf(-x)); }

// ---------------------------------------------------------------------------
// Phase 0a: convert the embedding table to bf16 once (row layout preserved).
// One block per vocab row; thread t converts element t. Pure bandwidth sweep.
// ---------------------------------------------------------------------------
__global__ __launch_bounds__(256) void pack_emb(
    const float* __restrict__ emb, unsigned short* __restrict__ embbf)
{
    size_t i = (size_t)blockIdx.x * 256 + threadIdx.x;
    embbf[i] = f2bf(emb[i]);
}

// ---------------------------------------------------------------------------
// Phase 0b: pack Wk_f/Wk_b/Wr_f/Wr_b [256][1024] f32 -> bf16 B-fragment layout.
// B-fragment (32x16 bf16) per (nt,kt): lane holds K of column nt*16 + lane%16,
// e<8: kt*32 + (lane<16 ? 0 : 8) + e, e>=8: kt*32 + 16 + (lane<16 ? 0 : 8) + e-8. Packed so each
// lane's 16 bf16 (32B) are contiguous -> two global_load_b128 per fragment.
// ---------------------------------------------------------------------------
__global__ __launch_bounds__(256) void pack_weights(
    const float* __restrict__ WkF, const float* __restrict__ WkB,
    const float* __restrict__ WrF, const float* __restrict__ WrB,
    unsigned short* __restrict__ WkP, unsigned short* __restrict__ WrP)
{
    int gid  = blockIdx.x * blockDim.x + threadIdx.x;  // 4 * 262144
    int mat  = gid >> 18;
    int r    = gid & 0x3FFFF;
    int e    = r & 15;
    int lane = (r >> 4) & 31;
    int frag = r >> 9;                 // nt*8 + kt
    int nt = frag >> 3, kt = frag & 7;
    int k  = kt * 32 + ((lane < 16) ? 0 : 8) + ((e < 8) ? e : e + 8);
    int n  = nt * 16 + (lane & 15);
    const float* src = (mat == 0) ? WkF : (mat == 1) ? WkB : (mat == 2) ? WrF : WrB;
    unsigned short* dst = (mat < 2) ? (WkP + mat * FRAG_ELEMS)
                                    : (WrP + (mat - 2) * FRAG_ELEMS);
    dst[r] = f2bf(src[k * G_ + n]);
}

// ---------------------------------------------------------------------------
// Phase 1: xw[dir][t][b][:] = emb_bf[tok(dir,t,b)] @ Wk_dir + b_dir  (WMMA)
// Row index in M = t*32 + b (time-major); dir=1 rows use token[b][T-1-t] so
// phase 2 consumes xw[ts] directly at scan step ts.
// A-fragments: two b128 loads of pre-converted bf16, no VALU conversion.
// Each wave: 32x32 output (2mt x 2nt tiles, 8 k-steps -> 32 WMMAs).
// ---------------------------------------------------------------------------
__global__ __launch_bounds__(256) void gemm_xw(
    const int* __restrict__ tokens, const unsigned short* __restrict__ embbf,
    const unsigned short* __restrict__ WkP,
    const float* __restrict__ bF, const float* __restrict__ bB,
    float* __restrict__ xw)
{
    int lin = blockIdx.x;
    int dir = lin / (256 * 8);
    int rem = lin % (256 * 8);
    int mb  = rem >> 3;                  // 256 blocks of 64 rows
    int nb  = rem & 7;                   // 8 blocks of 128 cols
    int wave = threadIdx.x >> 5;
    int lane = threadIdx.x & 31;
    int Mbase = mb * 64 + (wave >> 2) * 32;
    int Nbase = nb * 128 + (wave & 3) * 32;

    const unsigned short* Wp = WkP + dir * FRAG_ELEMS;
    const float* bias = dir ? bB : bF;

    int tok[2];
#pragma unroll
    for (int mt = 0; mt < 2; ++mt) {
        int row = Mbase + mt * 16 + (lane & 15);
        int tt = row >> 5, bb = row & 31;
        int tcol = dir ? (T_ - 1 - tt) : tt;
        tok[mt] = tokens[bb * T_ + tcol];
    }
    int klo = (lane < 16) ? 0 : 8;       // A-fragment lane K offset

    v8f acc[2][2] = {};
#pragma unroll 2
    for (int kt = 0; kt < 8; ++kt) {
        v16bf a[2];
#pragma unroll
        for (int mt = 0; mt < 2; ++mt) {
            const unsigned short* ebase = embbf + (size_t)tok[mt] * E_ + kt * 32;
            uint4 lo = *(const uint4*)(ebase + klo);        // K = kbase+klo .. +7
            uint4 hi = *(const uint4*)(ebase + 16 + klo);   // K = kbase+16+klo .. +7
            v16u au;
            ((uint4*)&au)[0] = lo;
            ((uint4*)&au)[1] = hi;
            a[mt] = __builtin_bit_cast(v16bf, au);
        }
#pragma unroll
        for (int ntl = 0; ntl < 2; ++ntl) {
            int nt = (Nbase >> 4) + ntl;
            v16u bu = *(const v16u*)(Wp + (((nt << 3) + kt) * 32 + lane) * 16);
            v16bf b = __builtin_bit_cast(v16bf, bu);
#pragma unroll
            for (int mt = 0; mt < 2; ++mt)
                acc[mt][ntl] = __builtin_amdgcn_wmma_f32_16x16x32_bf16(
                    false, a[mt], false, b, (short)0, acc[mt][ntl], false, false);
        }
    }

    float* out = xw + (size_t)dir * MROWS * G_;
#pragma unroll
    for (int mt = 0; mt < 2; ++mt) {
        int rowb = Mbase + mt * 16 + ((lane < 16) ? 0 : 8);
#pragma unroll
        for (int ntl = 0; ntl < 2; ++ntl) {
            int col = Nbase + ntl * 16 + (lane & 15);
            float bv = bias[col];
#pragma unroll
            for (int r = 0; r < 8; ++r)
                out[(size_t)(rowb + r) * G_ + col] = acc[mt][ntl][r] + bv;
        }
    }
}

// ---------------------------------------------------------------------------
// Phase 2: the sequential scan. One workgroup (1024 thr = 32 waves) per dir.
// LDS: z[32][1024] f32 (128K) + c[32][256] f32 (32K) + h[32][256] f32 (32K)
//      + hA bf16 A-fragment image of h (16K)  => 208K of the WGP's 320K.
// Per step: each wave does a 32x32 tile of h@Wr (8 kt, 32 WMMAs); Wr streams
// from L2 in packed-fragment form; gates/state update in f32; h scattered
// back into A-fragment LDS layout for the next step.
// ---------------------------------------------------------------------------
__global__ __launch_bounds__(1024) void lstm_seq(
    const int* __restrict__ tokens,
    const unsigned short* __restrict__ WrP,
    const float* __restrict__ xw,
    float* __restrict__ ys, float* __restrict__ sh, float* __restrict__ sc)
{
    int dir = blockIdx.x;
    extern __shared__ char smem[];
    float* zb = (float*)smem;                               // [32][1024]
    float* cS = zb + 32 * G_;                               // [32][256]
    float* hS = cS + B_ * U_;                               // [32][256]
    unsigned short* hA = (unsigned short*)(hS + B_ * U_);   // [2][8][32][16]

    int tid = threadIdx.x;
    for (int i = tid; i < B_ * U_; i += 1024) { cS[i] = 0.f; hS[i] = 0.f; }
    for (int i = tid; i < 2 * 8 * 32 * 16; i += 1024) hA[i] = 0;
    __syncthreads();

    int wave = tid >> 5, lane = tid & 31;
    const unsigned short* Wp = WrP + dir * FRAG_ELEMS;
    const float* xwd = xw + (size_t)dir * MROWS * G_;

    for (int ts = 0; ts < T_; ++ts) {
        // ---- z = h @ Wr via WMMA (uniform flow: EXEC all-1s) ----
        v8f acc[2][2] = {};
#pragma unroll 2
        for (int kt = 0; kt < 8; ++kt) {
            v16bf a[2];
#pragma unroll
            for (int mt = 0; mt < 2; ++mt) {
                v16u au = *(const v16u*)(hA + (((mt << 3) + kt) * 32 + lane) * 16);
                a[mt] = __builtin_bit_cast(v16bf, au);
            }
#pragma unroll
            for (int ntl = 0; ntl < 2; ++ntl) {
                int nt = wave * 2 + ntl;
                v16u bu = *(const v16u*)(Wp + (((nt << 3) + kt) * 32 + lane) * 16);
                v16bf b = __builtin_bit_cast(v16bf, bu);
#pragma unroll
                for (int mt = 0; mt < 2; ++mt)
                    acc[mt][ntl] = __builtin_amdgcn_wmma_f32_16x16x32_bf16(
                        false, a[mt], false, b, (short)0, acc[mt][ntl], false, false);
            }
        }
#pragma unroll
        for (int mt = 0; mt < 2; ++mt) {
            int rowb = mt * 16 + ((lane < 16) ? 0 : 8);
#pragma unroll
            for (int ntl = 0; ntl < 2; ++ntl) {
                int col = (wave * 2 + ntl) * 16 + (lane & 15);
#pragma unroll
                for (int r = 0; r < 8; ++r)
                    zb[(rowb + r) * G_ + col] = acc[mt][ntl][r];
            }
        }
        __syncthreads();

        // ---- gates + state update (f32), 8 (b,u) pairs per thread ----
        int torig = dir ? (T_ - 1 - ts) : ts;
        const float* zx = xwd + (size_t)ts * (B_ * G_);
#pragma unroll
        for (int p = 0; p < 8; ++p) {
            int idx = tid + (p << 10);
            int b = idx >> 8, u = idx & 255;
            const float* zrow = zb + b * G_;
            const float* xrow = zx + b * G_;
            float ig = sigmoidf(zrow[u]        + xrow[u]);
            float fg = sigmoidf(zrow[256 + u]  + xrow[256 + u]);
            float gg = fmaxf   (zrow[512 + u]  + xrow[512 + u], 0.f);
            float og = sigmoidf(zrow[768 + u]  + xrow[768 + u]);
            float cold = cS[b * U_ + u];
            float cnew = fg * cold + ig * gg;
            float hnew = og * fmaxf(cnew, 0.f);
            bool  m  = tokens[b * T_ + torig] != 0;
            float h2 = m ? hnew : hS[b * U_ + u];
            float c2 = m ? cnew : cold;
            cS[b * U_ + u] = c2;
            hS[b * U_ + u] = h2;
            ys[(size_t)b * (T_ * 2 * U_) + (size_t)torig * (2 * U_) + dir * U_ + u] = h2;
            // scatter h2 (bf16) into WMMA A-fragment layout
            int mt = b >> 4, r2 = b & 15, kt = u >> 5, ko = u & 31;
            int lane2, e;
            if      (ko < 8)  { lane2 = r2;      e = ko;      }
            else if (ko < 16) { lane2 = r2 + 16; e = ko - 8;  }
            else if (ko < 24) { lane2 = r2;      e = ko - 8;  }
            else              { lane2 = r2 + 16; e = ko - 16; }
            hA[(((mt << 3) + kt) * 32 + lane2) * 16 + e] = f2bf(h2);
        }
        __syncthreads();
    }

    // final states: state_h = [hf, cb], state_c = [cf, cb] (reference quirk)
    for (int i = tid; i < B_ * U_; i += 1024) {
        int b = i >> 8, u = i & 255;
        if (dir == 0) {
            sh[b * (2 * U_) + u] = hS[i];
            sc[b * (2 * U_) + u] = cS[i];
        } else {
            sh[b * (2 * U_) + U_ + u] = cS[i];
            sc[b * (2 * U_) + U_ + u] = cS[i];
        }
    }
}

// ---------------------------------------------------------------------------
extern "C" void kernel_launch(void* const* d_in, const int* in_sizes, int n_in,
                              void* d_out, int out_size, void* d_ws, size_t ws_size,
                              hipStream_t stream) {
    (void)in_sizes; (void)n_in; (void)out_size; (void)ws_size;
    const int*   tokens = (const int*)  d_in[0];
    const float* emb    = (const float*)d_in[1];
    const float* WkF    = (const float*)d_in[2];
    const float* WrF    = (const float*)d_in[3];
    const float* bF     = (const float*)d_in[4];
    const float* WkB    = (const float*)d_in[5];
    const float* WrB    = (const float*)d_in[6];
    const float* bB     = (const float*)d_in[7];
    float* out = (float*)d_out;
    char*  ws  = (char*)d_ws;

    // workspace layout:
    //   xw     : 2*16384*1024 f32                = 128 MB
    //   WkP    : 2*262144 bf16                   = 1 MB
    //   WrP    : 2*262144 bf16                   = 1 MB
    //   emb_bf : 50257*256 bf16                  = ~24.6 MB
    float* xw = (float*)ws;
    unsigned short* WkP   = (unsigned short*)(ws + (size_t)2 * MROWS * G_ * sizeof(float));
    unsigned short* WrP   = WkP + 2 * FRAG_ELEMS;
    unsigned short* embbf = WrP + 2 * FRAG_ELEMS;

    pack_emb<<<V_, 256, 0, stream>>>(emb, embbf);
    pack_weights<<<(4 * FRAG_ELEMS) / 256, 256, 0, stream>>>(WkF, WkB, WrF, WrB, WkP, WrP);
    gemm_xw<<<2 * 256 * 8, 256, 0, stream>>>(tokens, embbf, WkP, bF, bB, xw);

    float* ys = out;
    float* sh = out + (size_t)B_ * T_ * 2 * U_;
    float* sc = sh + (size_t)B_ * 2 * U_;
    size_t shmem = (32 * G_ + 2 * B_ * U_) * sizeof(float)
                 + (size_t)2 * 8 * 32 * 16 * sizeof(unsigned short);   // 212992 B
    hipFuncSetAttribute((const void*)lstm_seq,
                        hipFuncAttributeMaxDynamicSharedMemorySize, (int)shmem);
    lstm_seq<<<2, 1024, shmem, stream>>>(tokens, WrP, xw, ys, sh, sc);
}
